// NonlocalBlockCherel_67731634258705
// MI455X (gfx1250) — matrix operand order repaired and checked by reading, not compiled, not run on hardware
//
#include <hip/hip_runtime.h>
#include <hip/hip_bf16.h>
#include <math.h>
#include <stdint.h>

// ---------------- problem constants ----------------
#define BB  8
#define CC  32
#define KK  8
#define HH  128
#define WW  128
#define CKD 256                      // C*K
#define NPX (BB * HH * WW)           // 131072 pixels

typedef _Float16 v4h  __attribute__((ext_vector_type(4)));
typedef _Float16 v8h  __attribute__((ext_vector_type(8)));
typedef _Float16 v16h __attribute__((ext_vector_type(16)));
typedef float    v8f  __attribute__((ext_vector_type(8)));

typedef unsigned int u32x4 __attribute__((ext_vector_type(4)));
typedef int          i32x4 __attribute__((ext_vector_type(4)));
typedef int          i32x8 __attribute__((ext_vector_type(8)));

#if __has_builtin(__builtin_amdgcn_tensor_load_to_lds)
#define USE_TDM 1
#else
#define USE_TDM 0
#endif

// =====================================================================
// fp32 -> fp16 weight conversion
// =====================================================================
__global__ __launch_bounds__(256) void k_f32_to_f16(const float* __restrict__ s,
                                                    _Float16* __restrict__ d, int n) {
    int i = blockIdx.x * 256 + threadIdx.x;
    if (i < n) d[i] = (_Float16)s[i];
}

// =====================================================================
// Patch match + top-K gather.  x: NCHW f32 -> xb: [pixel][c*K+k] f16
// One block = one 16x16 tile of one batch image. 256 threads.
// =====================================================================
#define TS   16
#define HALO 3
#define XT   (TS + 2 * HALO)   // 22
#define DREG 18                // tile + patch halo 1

__global__ __launch_bounds__(256) void k_patch_match(const float* __restrict__ x,
                                                     _Float16* __restrict__ xb) {
    __shared__ float xtl[CC * XT * XT];   // 61952 B
    __shared__ float dbuf[DREG * DREG];   // 1296 B

    const int tid = threadIdx.x;
    const int b   = blockIdx.x >> 6;
    const int t   = blockIdx.x & 63;
    const int ty0 = (t >> 3) * TS;
    const int tx0 = (t & 7) * TS;

    // ---- load x tile + halo (edge-clamped => clipped shifts come free) ----
    for (int i = tid; i < CC * XT * XT; i += 256) {
        int c  = i / (XT * XT);
        int p  = i % (XT * XT);
        int ly = p / XT, lx = p % XT;
        int gy = ty0 + ly - HALO; gy = gy < 0 ? 0 : (gy > HH - 1 ? HH - 1 : gy);
        int gx = tx0 + lx - HALO; gx = gx < 0 ? 0 : (gx > WW - 1 ? WW - 1 : gx);
        xtl[i] = x[(((size_t)b * CC + c) * HH + gy) * WW + gx];
    }
    __syncthreads();

    const int ly = tid >> 4, lx = tid & 15;      // my pixel in tile
    const int py = ly + HALO, px = lx + HALO;    // my pixel in xtl coords

    float cost[25];

    // ---- 25 offsets: SSD map in LDS, 3x3 box-sum into registers ----
#pragma unroll
    for (int j = 0; j < 25; ++j) {
        const int dy = j / 5 - 2;
        const int dx = j % 5 - 2;
        // d over (tile + 1) region, zero outside image (conv zero-pad on d)
#pragma unroll
        for (int i = tid; i < DREG * DREG; i += 256) {
            int ry = i / DREG, rx = i % DREG;
            int gy = ty0 + ry - 1, gx = tx0 + rx - 1;
            float s = 0.0f;
            if (gy >= 0 && gy < HH && gx >= 0 && gx < WW) {
                int p0 = (ry + 2) * XT + (rx + 2);
                int p1 = (ry + 2 + dy) * XT + (rx + 2 + dx);
#pragma unroll
                for (int c = 0; c < CC; ++c) {
                    float dv = xtl[c * XT * XT + p0] - xtl[c * XT * XT + p1];
                    s += dv * dv;
                }
            }
            dbuf[i] = s;
        }
        __syncthreads();
        float acc = 0.0f;
#pragma unroll
        for (int u = 0; u < 3; ++u)
#pragma unroll
            for (int v = 0; v < 3; ++v)
                acc += dbuf[(ly + u) * DREG + (lx + v)];
        cost[j] = acc;
        __syncthreads();
    }

    // ---- top-K (lowest cost, lowest index on ties), fully unrolled ----
    int sel[KK];
    unsigned chosen = 0;
#pragma unroll
    for (int k = 0; k < KK; ++k) {
        float best = 3.4e38f; int bi = 0;
#pragma unroll
        for (int j = 0; j < 25; ++j) {
            bool free_j = ((chosen >> j) & 1u) == 0u;
            if (free_j && cost[j] < best) { best = cost[j]; bi = j; }
        }
        chosen |= (1u << bi);
        sel[k] = bi;
    }

    int offy[KK], offx[KK];
#pragma unroll
    for (int k = 0; k < KK; ++k) { offy[k] = sel[k] / 5 - 2; offx[k] = sel[k] % 5 - 2; }

    // ---- gather: per channel the K values are contiguous -> 16B stores ----
    const int gy = ty0 + ly, gx = tx0 + lx;
    _Float16* dst = xb + ((size_t)((b * HH + gy) * WW + gx)) * CKD;
#pragma unroll
    for (int c = 0; c < CC; ++c) {
        v8h v;
#pragma unroll
        for (int k = 0; k < KK; ++k)
            v[k] = (_Float16)xtl[c * XT * XT + (py + offy[k]) * XT + (px + offx[k])];
        *(v8h*)(dst + c * KK) = v;
    }
}

// =====================================================================
// Grouped KxK conv (dct): out[p][c*8+ko] = sum_ki w[c*8+ko][ki]*in[p][c*8+ki]
// =====================================================================
__global__ __launch_bounds__(256) void k_group8(const _Float16* __restrict__ in,
                                                const float* __restrict__ w,
                                                _Float16* __restrict__ out) {
    const int pix = blockIdx.x;
    const int co  = threadIdx.x;
    const _Float16* row = in + (size_t)pix * CKD + (co >> 3) * 8;
    float acc = 0.0f;
#pragma unroll
    for (int ki = 0; ki < 8; ++ki) acc += (float)row[ki] * w[co * 8 + ki];
    out[(size_t)pix * CKD + co] = (_Float16)acc;
}

// =====================================================================
// WMMA GEMM: Out[N,256] = A[N,256] * Wt[256o,256i]^T   (f16 in, f32 acc, f16 out)
// 128x128 tile, BK=32, 8 waves in 4(M) x 2(N), wave tile 32x64.
// A/B tiles staged by the Tensor Data Mover (double buffered); TDM pad
// config reproduces the LDT=40 (half) padded row stride in hardware.
// =====================================================================
#define GBM 128
#define GBN 128
#define GBK 32
#define LDT 40   // padded halves/row (kills ds_load_b128 bank conflicts)

#if USE_TDM
// 2D tile load: 128 rows x 32 halves from a [tensor_dim1, 256] f16 tensor.
__device__ __forceinline__ void tdm_load_2d(const _Float16* gtile,
                                            unsigned lds_byte_off,
                                            unsigned tensor_dim1) {
    unsigned long long ga = (unsigned long long)(size_t)gtile;
    u32x4 g0;
    g0[0] = 1u;                                     // count=1 (valid user D#)
    g0[1] = lds_byte_off;                           // lds_addr [63:32]
    g0[2] = (unsigned)(ga & 0xffffffffull);         // global_addr [95:64]
    g0[3] = (unsigned)((ga >> 32) & 0x01ffffffull)  // global_addr [120:96]
          | (2u << 30);                             // type=2 ("image") [127:126]
    i32x8 g1;
    g1[0] = (1 << 16)      // data_size = 2 bytes
          | (1 << 20)      // pad_enable
          | (3 << 22)      // pad_interval: 16 DWORDs (one 32-half row)
          | (3 << 25);     // pad_amount:   4 DWORDs -> 40-half row stride
    g1[1] = (int)(256u << 16);                      // tensor_dim0 = 256 (lo16)
    g1[2] = (int)((tensor_dim1 & 0xffffu) << 16);   // dim0 hi16=0 | dim1 lo16
    g1[3] = (int)((tensor_dim1 >> 16) & 0xffffu)    // dim1 hi16
          | (int)(32u << 16);                       // tile_dim0 = 32
    g1[4] = 128;                                    // tile_dim1=128, tile_dim2=0
    g1[5] = 256;                                    // tensor_dim0_stride lo32
    g1[6] = 0;                                      // stride0 hi16 | stride1 lo16
    g1[7] = 0;                                      // stride1 hi32
    i32x4 gz = {0, 0, 0, 0};
#if defined(__clang_major__) && (__clang_major__ >= 23)
    i32x8 gz8 = {0, 0, 0, 0, 0, 0, 0, 0};
    __builtin_amdgcn_tensor_load_to_lds(g0, g1, gz, gz, gz8, 0);
#else
    __builtin_amdgcn_tensor_load_to_lds(g0, g1, gz, gz, 0);
#endif
}
#endif

__device__ __forceinline__ v16h ld_frag16(const _Float16* p0, const _Float16* p1) {
    v8h lo = *(const v8h*)p0;
    v8h hi = *(const v8h*)p1;
    v16h r;
#pragma unroll
    for (int i = 0; i < 8; ++i) { r[i] = lo[i]; r[i + 8] = hi[i]; }
    return r;
}

__global__ __launch_bounds__(256) void k_gemm256(const _Float16* __restrict__ A,
                                                 const _Float16* __restrict__ Wt,
                                                 _Float16* __restrict__ Out) {
#if USE_TDM
    __shared__ __align__(16) _Float16 Alds[2][GBM * LDT];   // 2 x 10240 B
    __shared__ __align__(16) _Float16 Blds[2][GBN * LDT];
#else
    __shared__ __align__(16) _Float16 Alds[1][GBM * LDT];
    __shared__ __align__(16) _Float16 Blds[1][GBN * LDT];
#endif

    const int tid   = threadIdx.x;
    const int lane  = tid & 31;
    const int wave  = tid >> 5;
    const int waveM = wave >> 1;       // 0..3
    const int waveN = wave & 1;        // 0..1
    const int lh    = lane >> 4;       // lane half
    const int ll    = lane & 15;
    const int m0    = blockIdx.x * GBM;
    const int n0    = blockIdx.y * GBN;

    v8f acc[2][4];
    const v8f zero = {0.f, 0.f, 0.f, 0.f, 0.f, 0.f, 0.f, 0.f};
#pragma unroll
    for (int mi = 0; mi < 2; ++mi)
#pragma unroll
        for (int ni = 0; ni < 4; ++ni) acc[mi][ni] = zero;

#if USE_TDM
    if (wave == 0) {   // TDM is a wave-level DMA: issue once (uniform branch)
        tdm_load_2d(A  + (size_t)m0 * CKD, (unsigned)(size_t)&Alds[0][0], (unsigned)NPX);
        tdm_load_2d(Wt + (size_t)n0 * CKD, (unsigned)(size_t)&Blds[0][0], 256u);
    }
#endif

    for (int t = 0; t < CKD / GBK; ++t) {
        const int kt = t * GBK;
#if USE_TDM
        const int cur = t & 1;
        if (wave == 0) {
            if (t < CKD / GBK - 1) {
                // prefetch next tiles into the other buffer, then wait only
                // for the two oldest (current-buffer) DMAs -> overlap
                tdm_load_2d(A  + (size_t)m0 * CKD + (kt + GBK),
                            (unsigned)(size_t)&Alds[cur ^ 1][0], (unsigned)NPX);
                tdm_load_2d(Wt + (size_t)n0 * CKD + (kt + GBK),
                            (unsigned)(size_t)&Blds[cur ^ 1][0], 256u);
                __builtin_amdgcn_s_wait_tensorcnt(2);
            } else {
                __builtin_amdgcn_s_wait_tensorcnt(0);
            }
        }
        __syncthreads();
#else
        const int cur = 0;
#pragma unroll
        for (int i = tid; i < GBM * 4; i += 256) {   // 512 x uint4 per tile
            int row = i >> 2, c4 = i & 3;
            uint4 va = *(const uint4*)(A  + (size_t)(m0 + row) * CKD + kt + c4 * 8);
            *(uint4*)(&Alds[cur][row * LDT + c4 * 8]) = va;
            uint4 vb = *(const uint4*)(Wt + (size_t)(n0 + row) * CKD + kt + c4 * 8);
            *(uint4*)(&Blds[cur][row * LDT + c4 * 8]) = vb;
        }
        __syncthreads();
#endif

        // ---- fragments (2 x ds_load_b128 each) ----
        v16h afr[2], bfr[4];
#pragma unroll
        for (int mi = 0; mi < 2; ++mi) {
            int row = waveM * 32 + mi * 16 + ll;
            afr[mi] = ld_frag16(&Alds[cur][row * LDT + 8 * lh],
                                &Alds[cur][row * LDT + 16 + 8 * lh]);
        }
#pragma unroll
        for (int ni = 0; ni < 4; ++ni) {
            int co = waveN * 64 + ni * 16 + ll;
            bfr[ni] = ld_frag16(&Blds[cur][co * LDT + 8 * lh],
                                &Blds[cur][co * LDT + 16 + 8 * lh]);
        }
#pragma unroll
        for (int mi = 0; mi < 2; ++mi)
#pragma unroll
            for (int ni = 0; ni < 4; ++ni)
                acc[mi][ni] = __builtin_amdgcn_wmma_f32_16x16x32_f16(
                    false, afr[mi], false, bfr[ni], (short)0, acc[mi][ni], false, false);
        __syncthreads();   // all reads of [cur] done before it is re-staged
    }

    // ---- store D (lane ll = col, vgpr r = row 8*lh + r) ----
#pragma unroll
    for (int mi = 0; mi < 2; ++mi)
#pragma unroll
        for (int ni = 0; ni < 4; ++ni) {
            int gn = n0 + waveN * 64 + ni * 16 + ll;
#pragma unroll
            for (int r = 0; r < 8; ++r) {
                int gm = m0 + waveM * 32 + mi * 16 + 8 * lh + r;
                Out[(size_t)gm * CKD + gn] = (_Float16)acc[mi][ni][r];
            }
        }
}

// =====================================================================
// Depthwise 3x3 (zero pad) + activation. act: 0 = relu, 1 = sigmoid
// =====================================================================
__global__ __launch_bounds__(256) void k_dw3x3(const _Float16* __restrict__ in,
                                               const float* __restrict__ w,
                                               _Float16* __restrict__ out, int act) {
    __shared__ float wsm[CKD * 9];
    for (int i = threadIdx.x; i < CKD * 9; i += 256) wsm[i] = w[i];
    __syncthreads();

    const int pix = blockIdx.x * 4 + (threadIdx.x >> 6);
    const int c4  = (threadIdx.x & 63) * 4;
    const int bb  = pix >> 14;
    const int y   = (pix >> 7) & 127;
    const int x   = pix & 127;

    float acc0 = 0.f, acc1 = 0.f, acc2 = 0.f, acc3 = 0.f;
#pragma unroll
    for (int u = 0; u < 3; ++u)
#pragma unroll
        for (int v = 0; v < 3; ++v) {
            int yy = y + u - 1, xx = x + v - 1;
            if (yy >= 0 && yy < HH && xx >= 0 && xx < WW) {
                size_t idx = ((size_t)((bb << 14) + (yy << 7) + xx)) * CKD + c4;
                v4h d = *(const v4h*)(in + idx);
                int tap = u * 3 + v;
                acc0 += (float)d[0] * wsm[(c4 + 0) * 9 + tap];
                acc1 += (float)d[1] * wsm[(c4 + 1) * 9 + tap];
                acc2 += (float)d[2] * wsm[(c4 + 2) * 9 + tap];
                acc3 += (float)d[3] * wsm[(c4 + 3) * 9 + tap];
            }
        }
    float r0, r1, r2, r3;
    if (act == 0) {
        r0 = acc0 > 0.f ? acc0 : 0.f; r1 = acc1 > 0.f ? acc1 : 0.f;
        r2 = acc2 > 0.f ? acc2 : 0.f; r3 = acc3 > 0.f ? acc3 : 0.f;
    } else {
        r0 = 1.0f / (1.0f + __expf(-acc0)); r1 = 1.0f / (1.0f + __expf(-acc1));
        r2 = 1.0f / (1.0f + __expf(-acc2)); r3 = 1.0f / (1.0f + __expf(-acc3));
    }
    v4h o; o[0] = (_Float16)r0; o[1] = (_Float16)r1; o[2] = (_Float16)r2; o[3] = (_Float16)r3;
    *(v4h*)(out + (size_t)pix * CKD + c4) = o;
}

// =====================================================================
// filt = inv_dct( wiener * dct ), grouped 8x8 with fused elementwise mul
// =====================================================================
__global__ __launch_bounds__(256) void k_invdct_mul(const _Float16* __restrict__ wiener,
                                                    const _Float16* __restrict__ dct,
                                                    const float* __restrict__ w,
                                                    _Float16* __restrict__ filt) {
    const int pix = blockIdx.x;
    const int co  = threadIdx.x;
    const int g   = (co >> 3) * 8;
    const _Float16* wr = wiener + (size_t)pix * CKD + g;
    const _Float16* dr = dct    + (size_t)pix * CKD + g;
    float acc = 0.0f;
#pragma unroll
    for (int ki = 0; ki < 8; ++ki)
        acc += ((float)wr[ki] * (float)dr[ki]) * w[co * 8 + ki];
    filt[(size_t)pix * CKD + co] = (_Float16)acc;
}

// =====================================================================
// Aggregation: out NCHW f32 [8,32,128,128] = filt[N,256] x agg_w[32,256]^T
// =====================================================================
__global__ __launch_bounds__(256) void k_agg(const _Float16* __restrict__ filt,
                                             const float* __restrict__ aggw,
                                             float* __restrict__ out) {
    const int pix = blockIdx.x * 8 + (threadIdx.x >> 5);
    const int co  = threadIdx.x & 31;
    const _Float16* row = filt + (size_t)pix * CKD;
    __builtin_prefetch(row, 0, 0);   // global_prefetch_b8
    float acc = 0.0f;
#pragma unroll 8
    for (int ch = 0; ch < CKD; ++ch) acc += (float)row[ch] * aggw[co * CKD + ch];
    const int bb = pix >> 14;
    const int yx = pix & 16383;
    out[(((size_t)bb * CC + co) << 14) + yx] = acc;
}

// =====================================================================
// host-side launch sequence
// =====================================================================
extern "C" void kernel_launch(void* const* d_in, const int* in_sizes, int n_in,
                              void* d_out, int out_size, void* d_ws, size_t ws_size,
                              hipStream_t stream) {
    (void)in_sizes; (void)n_in; (void)out_size; (void)ws_size;

    const float* x      = (const float*)d_in[0];
    const float* dct_w  = (const float*)d_in[1];
    const float* w1     = (const float*)d_in[2];
    const float* w2     = (const float*)d_in[3];
    const float* w3     = (const float*)d_in[4];
    const float* w4     = (const float*)d_in[5];
    const float* w5     = (const float*)d_in[6];
    const float* w6     = (const float*)d_in[7];
    const float* inv_w  = (const float*)d_in[8];
    const float* agg_w  = (const float*)d_in[9];
    float* out = (float*)d_out;

    char* ws = (char*)d_ws;
    const size_t SZ = (size_t)NPX * CKD * sizeof(_Float16);   // 64 MB
    _Float16* xb   = (_Float16*)(ws + 0 * SZ);
    _Float16* dct  = (_Float16*)(ws + 1 * SZ);
    _Float16* bufA = (_Float16*)(ws + 2 * SZ);
    _Float16* bufB = (_Float16*)(ws + 3 * SZ);
    _Float16* w1f  = (_Float16*)(ws + 4 * SZ);
    _Float16* w3f  = (_Float16*)(ws + 4 * SZ + 1 * 131072);
    _Float16* w5f  = (_Float16*)(ws + 4 * SZ + 2 * 131072);

    // weight conversion (w1/w3/w5 as [co][ci] f16 for the B-operand)
    k_f32_to_f16<<<256, 256, 0, stream>>>(w1, w1f, CKD * CKD);
    k_f32_to_f16<<<256, 256, 0, stream>>>(w3, w3f, CKD * CKD);
    k_f32_to_f16<<<256, 256, 0, stream>>>(w5, w5f, CKD * CKD);

    // patch match + gather -> xb (NHWC f16)
    k_patch_match<<<BB * (HH / TS) * (WW / TS), 256, 0, stream>>>(x, xb);

    // dct branch
    k_group8<<<NPX, 256, 0, stream>>>(xb, dct_w, dct);

    dim3 ggrid(NPX / GBM, CKD / GBN);   // (1024, 2)

    // stage 1: w1 GEMM -> dw(w2)+relu
    k_gemm256<<<ggrid, 256, 0, stream>>>(xb, w1f, bufA);
    k_dw3x3<<<NPX / 4, 256, 0, stream>>>(bufA, w2, bufB, 0);

    // stage 2: w3 GEMM -> dw(w4)+relu
    k_gemm256<<<ggrid, 256, 0, stream>>>(bufB, w3f, bufA);
    k_dw3x3<<<NPX / 4, 256, 0, stream>>>(bufA, w4, bufB, 0);

    // stage 3: w5 GEMM -> dw(w6)+sigmoid  => wiener in bufB
    k_gemm256<<<ggrid, 256, 0, stream>>>(bufB, w5f, bufA);
    k_dw3x3<<<NPX / 4, 256, 0, stream>>>(bufA, w6, bufB, 1);

    // filt = inv_dct(wiener * dct) -> bufA
    k_invdct_mul<<<NPX, 256, 0, stream>>>(bufB, dct, inv_w, bufA);

    // aggregate -> NCHW f32 output
    k_agg<<<NPX / 8, 256, 0, stream>>>(bufA, agg_w, out);
}
